// BiLSTM_GNN_RUL_40561671143904
// MI455X (gfx1250) — matrix operand order repaired and checked by reading, not compiled, not run on hardware
//
#include <hip/hip_runtime.h>
#include <hip/hip_bf16.h>
#include <math.h>

// ---------------- problem constants (fixed shapes from the reference) ---------------
#define NNODES 4096
#define TSTEPS 64
#define FIN    22
#define NEDGES 65536
#define EE     (NEDGES + NNODES)   // edges + self loops
#define HEADS  4
#define GH     64
#define GDIM   256                  // GAT output (concat heads)
#define LH     128                  // LSTM hidden per direction
#define BDIM   256                  // BiLSTM output
#define G4     512                  // 4*LH
#define FH     50

typedef __attribute__((ext_vector_type(16))) __bf16 v16bf;
typedef __attribute__((ext_vector_type(8)))  __bf16 v8bf;
typedef __attribute__((ext_vector_type(4)))  __bf16 v4bf;
typedef __attribute__((ext_vector_type(8)))  float  v8f;
typedef __attribute__((ext_vector_type(4)))  int    i32x4;

// CDNA5 async global->LDS path (guarded: falls back to load+ds_store if absent)
#if defined(__gfx1250__) && __has_builtin(__builtin_amdgcn_global_load_async_to_lds_b128) && \
    __has_builtin(__builtin_amdgcn_s_wait_asynccnt)
#define USE_ASYNC_LDS 1
typedef __attribute__((address_space(1))) i32x4 gi32x4;   // global (device) int4
typedef __attribute__((address_space(3))) i32x4 li32x4;   // LDS (shared) int4
#else
#define USE_ASYNC_LDS 0
#endif

// ===================================================================================
//        FAST PATH: double-buffered LDS-staged WMMA GEMM (guard-free shapes)
// Requires M%128==0, N%64==0, K%32==0.  Block = 256 thr (8 waves) -> 128x64 C tile.
// A: f32 activations -> 4 b128 loads in flight, then cvt to bf16 into LDS.
// Bt: weights PRE-converted to bf16 and PRE-transposed to [N][K] -> B tile staging is
//     a pure 16B async global->LDS copy.  WMMA fragments are contiguous in LDS.
// ===================================================================================
#define BM 128
#define BN 64
#define BK 32

__global__ void __launch_bounds__(256)
gemm_wmma_bf16_tiled(const float* __restrict__ A, const __bf16* __restrict__ Bt,
                     const float* __restrict__ bias, float* __restrict__ C,
                     int M, int N, int K, int act, int accum)
{
  __shared__ __align__(32) __bf16 As[2][BM][BK];   // 2 x 8 KB
  __shared__ __align__(32) __bf16 Bs[2][BN][BK];   // 2 x 4 KB

  const int tnCount = N / BN;
  const int tm   = blockIdx.x / tnCount;
  const int tn   = blockIdx.x % tnCount;
  const int row0 = tm * BM;
  const int col0 = tn * BN;

  const int tid  = threadIdx.x;
  const int wave = tid >> 5;
  const int lane = tid & 31;

  // this wave's 16x64 strip
  const int wrow0 = row0 + wave * 16;
  const int cn    = lane & 15;
  const int mofs  = (lane >> 4) << 3;          // C frag: lanes 16-31 hold M=8..15
  const int akb   = (lane >> 4) << 4;          // A frag k base: 0 or 16
  const int bkb   = (lane >> 4) << 4;          // B frag k base: 0 or 16
  const int am    = wave * 16 + (lane & 15);   // A frag row (local)

  // A tile: 4 float4 per thread; fixed (row,k) pairs
  const int ar[4] = { (tid + 0)   >> 3, (tid + 256) >> 3,
                      (tid + 512) >> 3, (tid + 768) >> 3 };
  const int ak[4] = { ((tid + 0)   & 7) << 2, ((tid + 256) & 7) << 2,
                      ((tid + 512) & 7) << 2, ((tid + 768) & 7) << 2 };
  // B tile copy mapping: 256 threads x 8 bf16 (16B) = 64x32 tile
  const int bn_ld = tid >> 2;                  // n row 0..63
  const int bk_ld = (tid & 3) << 3;            // k 0,8,16,24

  v8f acc[4];
#pragma unroll
  for (int j = 0; j < 4; ++j) {
    v8f z = {};
    if (accum) {
      const int n = col0 + (j << 4) + cn;
#pragma unroll
      for (int r = 0; r < 8; ++r)
        z[r] = C[(size_t)(wrow0 + mofs + r) * N + n];
    }
    acc[j] = z;
  }

  auto load_tiles = [&](int k0, int buf) {
    // ---- B 64x32 bf16, pre-transposed: straight 16B async copy per thread
    const __bf16* gp = &Bt[(size_t)(col0 + bn_ld) * K + k0 + bk_ld];
    __bf16*       lp = &Bs[buf][bn_ld][bk_ld];
#if USE_ASYNC_LDS
    __builtin_amdgcn_global_load_async_to_lds_b128(
        (gi32x4*)(void*)gp, (li32x4*)(void*)lp, 0, 0);
#else
    *(v8bf*)lp = *(const v8bf*)gp;
#endif
    // ---- A 128x32 f32: issue all 4 b128 loads, then convert + ds_store
    float4 f[4];
#pragma unroll
    for (int p = 0; p < 4; ++p)
      f[p] = *(const float4*)&A[(size_t)(row0 + ar[p]) * K + k0 + ak[p]];
#pragma unroll
    for (int p = 0; p < 4; ++p) {
      v4bf c4;
      c4[0] = (__bf16)f[p].x; c4[1] = (__bf16)f[p].y;
      c4[2] = (__bf16)f[p].z; c4[3] = (__bf16)f[p].w;
      *(v4bf*)&As[buf][ar[p]][ak[p]] = c4;     // ds_store_b64
    }
  };

  load_tiles(0, 0);
  int cur = 0;
  for (int k0 = 0; k0 < K; k0 += BK) {
#if USE_ASYNC_LDS
    __builtin_amdgcn_s_wait_asynccnt(0);
#endif
    __syncthreads();
    if (k0 + BK < K) load_tiles(k0 + BK, cur ^ 1);   // overlap next tile with math

    // ---- A fragment: two 16B contiguous LDS reads
    v8bf a_lo = *(const v8bf*)&As[cur][am][akb];
    v8bf a_hi = *(const v8bf*)&As[cur][am][akb + 8];
    v16bf a = __builtin_shufflevector(a_lo, a_hi,
                                      0, 1, 2, 3, 4, 5, 6, 7,
                                      8, 9, 10, 11, 12, 13, 14, 15);
    // ---- 4 B fragments, back-to-back WMMAs
    v16bf bfr[4];
#pragma unroll
    for (int j = 0; j < 4; ++j)
      bfr[j] = *(const v16bf*)&Bs[cur][(j << 4) + (lane & 15)][bkb];
#pragma unroll
    for (int j = 0; j < 4; ++j)
      acc[j] = __builtin_amdgcn_wmma_f32_16x16x32_bf16(
                   false, a, false, bfr[j], (short)0, acc[j], false, false);
    cur ^= 1;
  }

#pragma unroll
  for (int j = 0; j < 4; ++j) {
    const int n  = col0 + (j << 4) + cn;
    const float bv = bias ? bias[n] : 0.0f;
#pragma unroll
    for (int r = 0; r < 8; ++r) {
      float v = acc[j][r] + bv;
      if (act == 1) v = fmaxf(v, 0.0f);
      C[(size_t)(wrow0 + mofs + r) * N + n] = v;
    }
  }
}

// ---- one-time weight pre-processing: f32 [K,N] -> bf16 [N,K] (transposed) ----------
__global__ void convert_transpose_bf16(const float* __restrict__ B, __bf16* __restrict__ Bt,
                                       int K, int N)
{
  int i = blockIdx.x * blockDim.x + threadIdx.x;   // over K*N, output index n*K+k
  if (i >= K * N) return;
  int n = i / K, k = i - n * K;
  Bt[i] = (__bf16)B[(size_t)k * N + n];
}

// ===================================================================================
//          GENERIC PATH: guarded WMMA GEMM for ragged shapes (K=22, N=1100, ...)
// ===================================================================================
__device__ __forceinline__ v16bf load_a_frag(const float* __restrict__ A, int lda,
                                             int row0, int k0, int M, int K)
{
  const int lane = threadIdx.x & 31;
  const int m    = row0 + (lane & 15);
  const int kb   = (lane >> 4) << 4;
  const bool mok = (m < M);
  const float* Ar = A + (size_t)m * lda;
  v16bf a;
#pragma unroll
  for (int v = 0; v < 8; ++v) {
    int k = k0 + kb + (v << 1);
    float x0 = (mok && (k     < K)) ? Ar[k]     : 0.0f;
    float x1 = (mok && (k + 1 < K)) ? Ar[k + 1] : 0.0f;
    a[2 * v]     = (__bf16)x0;
    a[2 * v + 1] = (__bf16)x1;
  }
  return a;
}

__device__ __forceinline__ v16bf load_b_frag(const float* __restrict__ B, int ldb,
                                             int k0, int col0, int K, int Nc)
{
  const int lane = threadIdx.x & 31;
  const int n    = col0 + (lane & 15);
  const int kb   = (lane >> 4) << 4;
  const bool nok = (n < Nc);
  v16bf b;
#pragma unroll
  for (int v = 0; v < 8; ++v) {
    int k = k0 + kb + (v << 1);
    float x0 = (nok && (k     < K)) ? B[(size_t)k       * ldb + n] : 0.0f;
    float x1 = (nok && (k + 1 < K)) ? B[(size_t)(k + 1) * ldb + n] : 0.0f;
    b[2 * v]     = (__bf16)x0;
    b[2 * v + 1] = (__bf16)x1;
  }
  return b;
}

__global__ void __launch_bounds__(256)
gemm_wmma_bf16(const float* __restrict__ A, const float* __restrict__ B,
               const float* __restrict__ bias, float* __restrict__ C,
               int M, int N, int K, int act, int accum)
{
  const int wavesPerBlock = blockDim.x >> 5;
  const int wave   = blockIdx.x * wavesPerBlock + (threadIdx.x >> 5);
  const int tilesN = (N + 63) >> 6;
  const int tilesM = (M + 15) >> 4;
  if (wave >= tilesM * tilesN) return;
  const int tm   = wave / tilesN;
  const int tn   = wave % tilesN;
  const int row0 = tm << 4;
  const int colB = tn << 6;

  const int lane = threadIdx.x & 31;
  const int cn   = lane & 15;
  const int mofs = (lane >> 4) << 3;

  v8f acc[4];
#pragma unroll
  for (int j = 0; j < 4; ++j) {
    v8f z = {};
    if (accum) {
      const int n = colB + (j << 4) + cn;
      if (n < N) {
#pragma unroll
        for (int r = 0; r < 8; ++r) {
          int m = row0 + mofs + r;
          z[r] = (m < M) ? C[(size_t)m * N + n] : 0.0f;
        }
      }
    }
    acc[j] = z;
  }

  for (int k0 = 0; k0 < K; k0 += 32) {
    v16bf a = load_a_frag(A, K, row0, k0, M, K);
#pragma unroll
    for (int j = 0; j < 4; ++j) {
      const int c0 = colB + (j << 4);
      if (c0 < N) {
        v16bf b = load_b_frag(B, N, k0, c0, K, N);
        acc[j] = __builtin_amdgcn_wmma_f32_16x16x32_bf16(
                     false, a, false, b, (short)0, acc[j], false, false);
      }
    }
  }

#pragma unroll
  for (int j = 0; j < 4; ++j) {
    const int n = colB + (j << 4) + cn;
    if (n >= N) continue;
    const float bv = bias ? bias[n] : 0.0f;
#pragma unroll
    for (int r = 0; r < 8; ++r) {
      const int m = row0 + mofs + r;
      if (m >= M) continue;
      float v = acc[j][r] + bv;
      if (act == 1) v = fmaxf(v, 0.0f);
      C[(size_t)m * N + n] = v;
    }
  }
}

// ---------------- elementwise / GAT / LSTM kernels ----------------------------------
__global__ void fill_f32(float* __restrict__ p, float v, int n)
{
  int i = blockIdx.x * blockDim.x + threadIdx.x;
  if (i < n) p[i] = v;
}

__global__ void slice_x(const float* __restrict__ x, float* __restrict__ h0, int t)
{
  int i = blockIdx.x * blockDim.x + threadIdx.x;   // over N*F
  if (i >= NNODES * FIN) return;
  int n = i / FIN, f = i - n * FIN;
  h0[i] = x[(size_t)n * TSTEPS * FIN + (size_t)t * FIN + f];
}

__global__ void gat_scores(const float* __restrict__ g, const float* __restrict__ a_src,
                           const float* __restrict__ a_dst, float* __restrict__ ssrc,
                           float* __restrict__ sdst)
{
  int i = blockIdx.x * blockDim.x + threadIdx.x;   // over N*HEADS
  if (i >= NNODES * HEADS) return;
  int n = i >> 2, h = i & 3;
  const float* gp = g + (size_t)n * GDIM + h * GH;
  const float* as = a_src + h * GH;
  const float* ad = a_dst + h * GH;
  float s1 = 0.f, s2 = 0.f;
#pragma unroll 8
  for (int c = 0; c < GH; ++c) { float v = gp[c]; s1 += v * as[c]; s2 += v * ad[c]; }
  ssrc[i] = s1; sdst[i] = s2;
}

__device__ __forceinline__ void atomicMaxF(float* addr, float val)
{
  unsigned int* ua = (unsigned int*)addr;
  unsigned int cur = __float_as_uint(*addr);
  while (__uint_as_float(cur) < val) {
    unsigned int old = atomicCAS(ua, cur, __float_as_uint(val));
    if (old == cur) break;
    cur = old;
  }
}

__device__ __forceinline__ void edge_nodes(const int* __restrict__ ei, int e, int& s, int& d)
{
  if (e < NEDGES) { s = ei[e]; d = ei[NEDGES + e]; }
  else            { s = d = e - NEDGES; }          // self loops appended
}

__global__ void edge_logits_max(const int* __restrict__ ei, const float* __restrict__ ssrc,
                                const float* __restrict__ sdst, float* __restrict__ elog,
                                float* __restrict__ mmax)
{
  int i = blockIdx.x * blockDim.x + threadIdx.x;   // over EE*HEADS
  if (i >= EE * HEADS) return;
  int e = i >> 2, h = i & 3;
  int s, d; edge_nodes(ei, e, s, d);
  float v = ssrc[s * HEADS + h] + sdst[d * HEADS + h];
  v = (v > 0.f) ? v : 0.2f * v;                    // leaky_relu(0.2)
  elog[i] = v;
  atomicMaxF(&mmax[d * HEADS + h], v);
}

__global__ void edge_expsum(const int* __restrict__ ei, const float* __restrict__ elog,
                            const float* __restrict__ mmax, float* __restrict__ wbuf,
                            float* __restrict__ ssum)
{
  int i = blockIdx.x * blockDim.x + threadIdx.x;
  if (i >= EE * HEADS) return;
  int e = i >> 2, h = i & 3;
  int s, d; edge_nodes(ei, e, s, d);
  float w = __expf(elog[i] - mmax[d * HEADS + h]);
  wbuf[i] = w;
  atomicAdd(&ssum[d * HEADS + h], w);
}

__global__ void edge_aggregate(const int* __restrict__ ei, const float* __restrict__ g,
                               const float* __restrict__ wbuf, const float* __restrict__ ssum,
                               float* __restrict__ agg)
{
  long long i = (long long)blockIdx.x * blockDim.x + threadIdx.x;  // over EE*GDIM
  if (i >= (long long)EE * GDIM) return;
  int e = (int)(i >> 8);
  int c = (int)(i & 255);
  int h = c >> 6;
  int s, d; edge_nodes(ei, e, s, d);
  float alpha = wbuf[e * HEADS + h] / (ssum[d * HEADS + h] + 1e-16f);
  atomicAdd(&agg[(size_t)d * GDIM + c], g[(size_t)s * GDIM + c] * alpha);
}

__global__ void gat_finalize(const float* __restrict__ agg, const float* __restrict__ bias,
                             float* __restrict__ out)
{
  int i = blockIdx.x * blockDim.x + threadIdx.x;   // over N*GDIM
  if (i >= NNODES * GDIM) return;
  int c = i & 255;
  float v = agg[i] + bias[c];
  out[i] = (v > 0.f) ? v : (__expf(v) - 1.0f);     // ELU
}

__global__ void lstm_cell(const float* __restrict__ gates, const float* __restrict__ bih,
                          const float* __restrict__ bhh, float* __restrict__ h,
                          float* __restrict__ c, float* __restrict__ seq_out, int col0)
{
  int i = blockIdx.x * blockDim.x + threadIdx.x;   // over N*LH
  if (i >= NNODES * LH) return;
  int n = i >> 7, j = i & 127;
  const float* z = gates + (size_t)n * G4;
  float zi = z[j]          + bih[j]          + bhh[j];
  float zf = z[LH + j]     + bih[LH + j]     + bhh[LH + j];
  float zg = z[2 * LH + j] + bih[2 * LH + j] + bhh[2 * LH + j];
  float zo = z[3 * LH + j] + bih[3 * LH + j] + bhh[3 * LH + j];
  float si = 1.f / (1.f + __expf(-zi));
  float sf = 1.f / (1.f + __expf(-zf));
  float so = 1.f / (1.f + __expf(-zo));
  float cc = sf * c[i] + si * tanhf(zg);
  float hh = so * tanhf(cc);
  c[i] = cc; h[i] = hh;
  seq_out[(size_t)n * BDIM + col0 + j] = hh;
}

__global__ void softmax_rows(float* __restrict__ p, int rows, int cols)
{
  int n = blockIdx.x * blockDim.x + threadIdx.x;
  if (n >= rows) return;
  float* r = p + (size_t)n * cols;
  float m = -3.0e38f;
  for (int c = 0; c < cols; ++c) m = fmaxf(m, r[c]);
  float s = 0.f;
  for (int c = 0; c < cols; ++c) { float e = __expf(r[c] - m); r[c] = e; s += e; }
  float inv = 1.0f / s;
  for (int c = 0; c < cols; ++c) r[c] *= inv;
}

// ---------------- host-side helpers --------------------------------------------------
static inline void gemm_fast(const float* A, const __bf16* Bt, const float* bias, float* C,
                             int M, int N, int K, int act, int accum, hipStream_t s)
{
  int blocks = (M / BM) * (N / BN);
  gemm_wmma_bf16_tiled<<<blocks, 256, 0, s>>>(A, Bt, bias, C, M, N, K, act, accum);
}

static inline void gemm_ragged(const float* A, const float* B, const float* bias, float* C,
                               int M, int N, int K, int act, int accum, hipStream_t s)
{
  int tiles  = ((M + 15) >> 4) * ((N + 63) >> 6);
  int wpb    = 8;
  int blocks = (tiles + wpb - 1) / wpb;
  gemm_wmma_bf16<<<blocks, 32 * wpb, 0, s>>>(A, B, bias, C, M, N, K, act, accum);
}

extern "C" void kernel_launch(void* const* d_in, const int* in_sizes, int n_in,
                              void* d_out, int out_size, void* d_ws, size_t ws_size,
                              hipStream_t stream)
{
  (void)in_sizes; (void)n_in; (void)out_size; (void)ws_size;

  const float* x  = (const float*)d_in[0];
  const int*   ei = (const int*)d_in[1];
  const float* gW[2]  = {(const float*)d_in[2], (const float*)d_in[6]};
  const float* gAs[2] = {(const float*)d_in[3], (const float*)d_in[7]};
  const float* gAd[2] = {(const float*)d_in[4], (const float*)d_in[8]};
  const float* gB[2]  = {(const float*)d_in[5], (const float*)d_in[9]};
  const float *Wih[2][2], *Whh[2][2], *bih[2][2], *bhh[2][2];
  for (int l = 0; l < 2; ++l) {
    int base = 10 + l * 8;
    Wih[l][0] = (const float*)d_in[base + 0]; Whh[l][0] = (const float*)d_in[base + 1];
    bih[l][0] = (const float*)d_in[base + 2]; bhh[l][0] = (const float*)d_in[base + 3];
    Wih[l][1] = (const float*)d_in[base + 4]; Whh[l][1] = (const float*)d_in[base + 5];
    bih[l][1] = (const float*)d_in[base + 6]; bhh[l][1] = (const float*)d_in[base + 7];
  }
  const int hb = 26;
  const float* f1_w = (const float*)d_in[hb + 0]; const float* f1_b = (const float*)d_in[hb + 1];
  const float* f2_w = (const float*)d_in[hb + 2]; const float* f2_b = (const float*)d_in[hb + 3];
  const float* r1_w = (const float*)d_in[hb + 4]; const float* r1_b = (const float*)d_in[hb + 5];
  const float* r2_w = (const float*)d_in[hb + 6]; const float* r2_b = (const float*)d_in[hb + 7];
  const float* r3_w = (const float*)d_in[hb + 8]; const float* r3_b = (const float*)d_in[hb + 9];
  const float* a_w  = (const float*)d_in[hb + 10]; const float* a_b = (const float*)d_in[hb + 11];

  // ---- carve workspace ----
  char* wsp = (char*)d_ws;
  size_t off = 0;
  auto carve = [&](size_t bytes) -> void* {
    void* p = (void*)(wsp + off);
    off = (off + bytes + 255) & ~(size_t)255;
    return p;
  };
  float* seqA   = (float*)carve((size_t)TSTEPS * NNODES * BDIM * 4);
  float* seqB   = (float*)carve((size_t)TSTEPS * NNODES * BDIM * 4);
  float* gates  = (float*)carve((size_t)NNODES * G4 * 4);
  float* hstate = (float*)carve((size_t)NNODES * LH * 4);
  float* cstate = (float*)carve((size_t)NNODES * LH * 4);
  float* h0     = (float*)carve((size_t)NNODES * FIN * 4);
  float* h1     = (float*)carve((size_t)NNODES * GDIM * 4);
  float* gbuf   = (float*)carve((size_t)NNODES * GDIM * 4);
  float* agg    = (float*)carve((size_t)NNODES * GDIM * 4);
  float* ssrc   = (float*)carve((size_t)NNODES * HEADS * 4);
  float* sdst   = (float*)carve((size_t)NNODES * HEADS * 4);
  float* mmax   = (float*)carve((size_t)NNODES * HEADS * 4);
  float* ssum   = (float*)carve((size_t)NNODES * HEADS * 4);
  float* elog   = (float*)carve((size_t)EE * HEADS * 4);
  float* wbuf   = (float*)carve((size_t)EE * HEADS * 4);
  float* fbuf   = (float*)carve((size_t)NNODES * 256 * 4);
  float* rbuf1  = (float*)carve((size_t)NNODES * 128 * 4);
  float* rbuf2  = (float*)carve((size_t)NNODES * 64 * 4);
  // bf16 transposed weights for the fast path
  __bf16* gW1t  = (__bf16*)carve((size_t)GDIM * GDIM * 2);
  __bf16 *Wiht[2][2], *Whht[2][2];
  for (int l = 0; l < 2; ++l)
    for (int d = 0; d < 2; ++d) {
      Wiht[l][d] = (__bf16*)carve((size_t)BDIM * G4 * 2);
      Whht[l][d] = (__bf16*)carve((size_t)LH * G4 * 2);
    }
  __bf16* f1t = (__bf16*)carve((size_t)BDIM * 256 * 2);
  __bf16* r1t = (__bf16*)carve((size_t)BDIM * 128 * 2);
  __bf16* r2t = (__bf16*)carve((size_t)128 * 64 * 2);

  const int TPB = 256;
  #define GRID(n) (((n) + TPB - 1) / TPB)

  // ---- one-time weight pre-conversion (f32 [K,N] -> bf16 [N,K]) ----
  auto conv = [&](const float* B, __bf16* Bt, int K, int N) {
    convert_transpose_bf16<<<GRID(K * N), TPB, 0, stream>>>(B, Bt, K, N);
  };
  conv(gW[1], gW1t, GDIM, GDIM);
  for (int l = 0; l < 2; ++l)
    for (int d = 0; d < 2; ++d) {
      conv(Wih[l][d], Wiht[l][d], BDIM, G4);
      conv(Whh[l][d], Whht[l][d], LH, G4);
    }
  conv(f1_w, f1t, BDIM, 256);
  conv(r1_w, r1t, BDIM, 128);
  conv(r2_w, r2t, 128, 64);

  // ---- GAT over timesteps -> seqA[t] ----
  for (int t = 0; t < TSTEPS; ++t) {
    slice_x<<<GRID(NNODES * FIN), TPB, 0, stream>>>(x, h0, t);
    const float* hin = h0; int din = FIN;
    for (int l = 0; l < 2; ++l) {
      float* dst = (l == 0) ? h1 : (seqA + (size_t)t * NNODES * BDIM);
      if (l == 0) gemm_ragged(hin, gW[0], nullptr, gbuf, NNODES, GDIM, din, 0, 0, stream);
      else        gemm_fast(hin, gW1t, nullptr, gbuf, NNODES, GDIM, din, 0, 0, stream);
      gat_scores<<<GRID(NNODES * HEADS), TPB, 0, stream>>>(gbuf, gAs[l], gAd[l], ssrc, sdst);
      fill_f32<<<GRID(NNODES * HEADS), TPB, 0, stream>>>(mmax, -3.0e38f, NNODES * HEADS);
      fill_f32<<<GRID(NNODES * HEADS), TPB, 0, stream>>>(ssum, 0.f, NNODES * HEADS);
      fill_f32<<<GRID(NNODES * GDIM), TPB, 0, stream>>>(agg, 0.f, NNODES * GDIM);
      edge_logits_max<<<GRID(EE * HEADS), TPB, 0, stream>>>(ei, ssrc, sdst, elog, mmax);
      edge_expsum<<<GRID(EE * HEADS), TPB, 0, stream>>>(ei, elog, mmax, wbuf, ssum);
      long long tot = (long long)EE * GDIM;
      edge_aggregate<<<(int)((tot + TPB - 1) / TPB), TPB, 0, stream>>>(ei, gbuf, wbuf, ssum, agg);
      gat_finalize<<<GRID(NNODES * GDIM), TPB, 0, stream>>>(agg, gB[l], dst);
      hin = h1; din = GDIM;
    }
  }

  // ---- 2-layer BiLSTM (ping-pong seqA <-> seqB) ----
  const float* lin = seqA; float* lout = seqB;
  for (int l = 0; l < 2; ++l) {
    for (int dir = 0; dir < 2; ++dir) {
      fill_f32<<<GRID(NNODES * LH), TPB, 0, stream>>>(hstate, 0.f, NNODES * LH);
      fill_f32<<<GRID(NNODES * LH), TPB, 0, stream>>>(cstate, 0.f, NNODES * LH);
      for (int s = 0; s < TSTEPS; ++s) {
        int t = dir ? (TSTEPS - 1 - s) : s;
        const float* xt = lin + (size_t)t * NNODES * BDIM;
        gemm_fast(xt, Wiht[l][dir], nullptr, gates, NNODES, G4, BDIM, 0, 0, stream);
        gemm_fast(hstate, Whht[l][dir], nullptr, gates, NNODES, G4, LH, 0, 1, stream);
        lstm_cell<<<GRID(NNODES * LH), TPB, 0, stream>>>(
            gates, bih[l][dir], bhh[l][dir], hstate, cstate,
            lout + (size_t)t * NNODES * BDIM, dir ? LH : 0);
      }
    }
    const float* tmp = lin; lin = lout; lout = (float*)tmp;
  }
  const float* last = lin + (size_t)(TSTEPS - 1) * NNODES * BDIM;  // [N, 256]

  // ---- heads -> d_out = [forecast | rul | attn] ----
  float* out_fore = (float*)d_out;
  float* out_rul  = out_fore + (size_t)NNODES * FH * FIN;
  float* out_attn = out_rul + NNODES;

  gemm_fast(last,  f1t,  f1_b, fbuf,     NNODES, 256,      BDIM, 1, 0, stream);
  gemm_ragged(fbuf, f2_w, f2_b, out_fore, NNODES, FH * FIN, 256,  0, 0, stream);
  gemm_fast(last,  r1t,  r1_b, rbuf1,    NNODES, 128,      BDIM, 1, 0, stream);
  gemm_fast(rbuf1, r2t,  r2_b, rbuf2,    NNODES, 64,       128,  1, 0, stream);
  gemm_ragged(rbuf2, r3_w, r3_b, out_rul, NNODES, 1,        64,   0, 0, stream);
  gemm_ragged(last,  a_w,  a_b, out_attn, NNODES, FIN,      BDIM, 0, 0, stream);
  softmax_rows<<<GRID(NNODES), TPB, 0, stream>>>(out_attn, NNODES, FIN);
  #undef GRID
}
